// HirEncRNNDecoder_59751585021966
// MI455X (gfx1250) — compile-verified, matrix operand order checked by reading, not run on hardware
//
#include <hip/hip_runtime.h>

// ---------------- problem constants ----------------
constexpr int Bn = 128, Ln = 400, Sn = 30, Vn = 50000, EMBn = 128, Hn = 512, Mn = 512, OOVn = 30;
constexpr int VPn = Vn + OOVn; // 50030
constexpr int NTv = Vn / 16;   // 3125 vocab N-tiles

// ---------------- d_out layout (floats, reference return order) ----------------
constexpr size_t OUT_FINAL = 0;                                   // [B, V+OOV]
constexpr size_t OUT_HNEXT = (size_t)Bn * VPn;                    // [1,B,H]
constexpr size_t OUT_CTXW  = OUT_HNEXT + (size_t)Bn * Hn;         // [B,M]
constexpr size_t OUT_ATTNW = OUT_CTXW  + (size_t)Bn * Mn;         // [B,L]
constexpr size_t OUT_PGEN  = OUT_ATTNW + (size_t)Bn * Ln;         // [B,1]
constexpr size_t OUT_COVW  = OUT_PGEN  + (size_t)Bn;              // [B,L]

// ---------------- workspace layout (4-byte elements) ----------------
constexpr size_t WS_YEMB    = 0;
constexpr size_t WS_DPROJW  = WS_YEMB    + (size_t)Bn * EMBn;
constexpr size_t WS_DPROJS  = WS_DPROJW  + (size_t)Bn * Hn;
constexpr size_t WS_SCORESW = WS_DPROJS  + (size_t)Bn * Hn;
constexpr size_t WS_SCORESS = WS_SCORESW + (size_t)Bn * Ln;
constexpr size_t WS_CTXS    = WS_SCORESS + (size_t)Bn * Sn;
constexpr size_t WS_HID     = WS_CTXS    + (size_t)Bn * Hn;
constexpr size_t WS_LOGITS  = WS_HID     + (size_t)Bn * Hn;       // [B, V] f32
constexpr size_t WS_WMPK    = WS_LOGITS  + (size_t)Bn * Vn;       // w_mem_proj bf16-pair packed [256][512] u32
constexpr size_t WS_SMPK    = WS_WMPK    + (size_t)256 * Hn;      // s_mem_proj packed
constexpr size_t WS_VD2PK   = WS_SMPK    + (size_t)256 * Hn;      // vd2_w packed [256][50000] u32

// ---------------- WMMA helper types ----------------
typedef __attribute__((ext_vector_type(16))) __bf16 bfvec16;
typedef __attribute__((ext_vector_type(2)))  __bf16 bfvec2;
typedef __attribute__((ext_vector_type(8)))  float  fvec8;

#define FZ8 {0.f,0.f,0.f,0.f,0.f,0.f,0.f,0.f}
#define WMMA_BF16(a, b, c) \
  __builtin_amdgcn_wmma_f32_16x16x32_bf16(false, (a), false, (b), (short)0, (c), false, false)

// f32 pair -> packed bf16 pair (round-to-nearest-even), lo in [15:0]
__device__ __forceinline__ unsigned int pack2bf(float a, float b) {
  union { float f; unsigned u; } x, y; x.f = a; y.f = b;
  unsigned ra = (x.u + 0x7FFFu + ((x.u >> 16) & 1u)) >> 16;
  unsigned rb = (y.u + 0x7FFFu + ((y.u >> 16) & 1u)) & 0xFFFF0000u;
  return (ra & 0xFFFFu) | rb;
}

// A fragment (16x32 bf16, MxK) from LDS slab of packed pairs, row stride 257 u32
// (padded to avoid 16-way LDS bank conflicts). ISA 7.12.2 wave32 layout.
__device__ __forceinline__ bfvec16 load_a_packed(const unsigned int (*slab)[257],
                                                 int ro, int k0) {
  int lane = threadIdx.x & 31;
  int m  = ro + (lane & 15);
  int pb = (k0 >> 1) + ((lane >> 4) << 2);      // pair index base: kb/2 = half*4
  bfvec16 a;
#pragma unroll
  for (int j = 0; j < 8; ++j) {
    int p = pb + ((j >> 2) << 3) + (j & 3);
    bfvec2 t = __builtin_bit_cast(bfvec2, slab[m][p]);
    a[2*j] = t[0]; a[2*j+1] = t[1];
  }
  return a;
}

// B fragment (32x16 bf16, KxN): Wp is packed [K/2][N] u32 -> lanes sharing a k
// read 16 consecutive n (coalesced global_load_b32).
__device__ __forceinline__ bfvec16 load_b_packed(const unsigned int* __restrict__ Wp,
                                                 int n0, int k0, int N) {
  int lane = threadIdx.x & 31;
  int n  = n0 + (lane & 15);
  int pb = (k0 >> 1) + ((lane >> 4) << 2);
  bfvec16 v;
#pragma unroll
  for (int j = 0; j < 8; ++j) {
    int p = pb + ((j >> 2) << 3) + (j & 3);
    bfvec2 t = __builtin_bit_cast(bfvec2, Wp[(size_t)p * N + n]);
    v[2*j] = t[0]; v[2*j+1] = t[1];
  }
  return v;
}

// ===== kernel 0: pack f32 weights [N][512] -> bf16-pair u32 [256][N] =====
__global__ void pack512_kernel(const float* __restrict__ W, unsigned int* __restrict__ out,
                               int N) {
  size_t total = (size_t)N * 256;
  for (size_t i = (size_t)blockIdx.x * 256 + threadIdx.x; i < total;
       i += (size_t)gridDim.x * 256) {
    int n = (int)(i >> 8), p = (int)(i & 255);
    out[(size_t)p * N + n] = pack2bf(W[(size_t)n * 512 + 2 * p],
                                     W[(size_t)n * 512 + 2 * p + 1]);
  }
}

// ================= kernel 1: embedding gather + GRU step =================
__global__ void gru_kernel(const float* __restrict__ emb_table, const int* __restrict__ y,
                           const float* __restrict__ h,
                           const float* __restrict__ w_ih, const float* __restrict__ w_hh,
                           const float* __restrict__ b_ih, const float* __restrict__ b_hh,
                           float* __restrict__ h_next, float* __restrict__ y_emb_ws) {
  int b = blockIdx.x, tid = threadIdx.x;
  __shared__ float emb[EMBn];
  __shared__ float hp[Hn];
  int yi = y[b];
  if (tid < EMBn) {
    float e = emb_table[(size_t)yi * EMBn + tid];
    emb[tid] = e;
    y_emb_ws[b * EMBn + tid] = e;
  }
  hp[tid] = h[b * Hn + tid];
  hp[tid + 256] = h[b * Hn + tid + 256];
  __syncthreads();
  for (int j = tid; j < Hn; j += 256) {
    float ir = b_ih[j], iz = b_ih[Hn + j], inn = b_ih[2 * Hn + j];
    const float* wr = &w_ih[(size_t)j * EMBn];
    const float* wz = &w_ih[(size_t)(Hn + j) * EMBn];
    const float* wn = &w_ih[(size_t)(2 * Hn + j) * EMBn];
    for (int e = 0; e < EMBn; ++e) { float x = emb[e]; ir += x*wr[e]; iz += x*wz[e]; inn += x*wn[e]; }
    float hr = b_hh[j], hz = b_hh[Hn + j], hn = b_hh[2 * Hn + j];
    const float* ur = &w_hh[(size_t)j * Hn];
    const float* uz = &w_hh[(size_t)(Hn + j) * Hn];
    const float* un = &w_hh[(size_t)(2 * Hn + j) * Hn];
    for (int k = 0; k < Hn; ++k) { float x = hp[k]; hr += x*ur[k]; hz += x*uz[k]; hn += x*un[k]; }
    float r = 1.f / (1.f + __expf(-(ir + hr)));
    float z = 1.f / (1.f + __expf(-(iz + hz)));
    float n = tanhf(inn + r * hn);
    h_next[b * Hn + j] = (1.f - z) * n + z * hp[j];
  }
}

// ================= kernel 2: decoder projections (word + sent) =================
__global__ void decproj_kernel(const float* __restrict__ hlast,
                               const float* __restrict__ wW, const float* __restrict__ bW,
                               const float* __restrict__ wS, const float* __restrict__ bS,
                               float* __restrict__ dW, float* __restrict__ dS) {
  int b = blockIdx.x, tid = threadIdx.x;
  bool sent = (blockIdx.y != 0);
  const float* W  = sent ? wS : wW;
  const float* bb = sent ? bS : bW;
  float* out      = sent ? dS : dW;
  __shared__ float hl[Hn];
  hl[tid] = hlast[b * Hn + tid];
  hl[tid + 256] = hlast[b * Hn + tid + 256];
  __syncthreads();
  for (int j = tid; j < Hn; j += 256) {
    float acc = bb[j];
    const float* w = &W[(size_t)j * Hn];
    for (int k = 0; k < Hn; ++k) acc += hl[k] * w[k];
    out[b * Hn + j] = acc;
  }
}

// ======= kernel 3/4: attention score GEMM (bf16 WMMA) + tanh + v-dot =======
// 32 rows per block; each wave runs FOUR independent WMMA chains per inner step
// (2 A-tiles x 2 B-tiles) so back-to-back WMMAs never share an accumulator.
template<bool COVER>
__global__ void attn_score_wmma(const float* __restrict__ mem, int Ldim,
                                const unsigned int* __restrict__ Wp,  // packed [256][H]
                                const float* __restrict__ dproj,
                                const float* __restrict__ vvec,
                                const float* __restrict__ covp,
                                const float* __restrict__ cov,
                                float* __restrict__ scores) {
  int b = blockIdx.y;
  int l0 = blockIdx.x * 32;
  int tid = threadIdx.x, wave = tid >> 5, lane = tid & 31;
  __shared__ unsigned int slab[32][257];
  __shared__ float sscore[32];
  for (int i = tid; i < 32 * 256; i += 128) {
    int r = i >> 8, p = i & 255;
    int l = l0 + r;
    float x0 = 0.f, x1 = 0.f;
    if (l < Ldim) {
      const float* src = &mem[((size_t)b * Ldim + l) * Mn + 2 * p];
      x0 = src[0]; x1 = src[1];
    }
    slab[r][p] = pack2bf(x0, x1);
  }
  if (tid < 32) sscore[tid] = 0.f;
  __syncthreads();

  int half = lane >> 4;
  float acc[16];
#pragma unroll
  for (int r = 0; r < 16; ++r) acc[r] = 0.f;
  float cvt[16];
  if (COVER) {
#pragma unroll
    for (int r = 0; r < 16; ++r) {
      int m = ((r & 8) << 1) + (r & 7) + (half << 3);   // tile (r>=8) rows +16
      int l = l0 + m;
      cvt[r] = (l < Ldim) ? cov[b * Ldim + l] : 0.f;
    }
  }

  for (int t = 0; t < 8; t += 2) {           // wave covers 8 column tiles, 2 at a time
    int d0 = ((wave << 3) + t) << 4;
    int d1 = d0 + 16;
    fvec8 c00 = FZ8, c10 = FZ8, c01 = FZ8, c11 = FZ8;
#pragma unroll 2
    for (int kc = 0; kc < 16; ++kc) {
      bfvec16 b0 = load_b_packed(Wp, d0, kc * 32, Hn);
      bfvec16 b1 = load_b_packed(Wp, d1, kc * 32, Hn);
      bfvec16 a0 = load_a_packed(slab, 0,  kc * 32);
      bfvec16 a1 = load_a_packed(slab, 16, kc * 32);
      c00 = WMMA_BF16(a0, b0, c00);
      c10 = WMMA_BF16(a1, b0, c10);
      c01 = WMMA_BF16(a0, b1, c01);
      c11 = WMMA_BF16(a1, b1, c11);
    }
    int n0 = d0 + (lane & 15), n1 = d1 + (lane & 15);
    float dp0 = dproj[b * Hn + n0], dp1 = dproj[b * Hn + n1];
    float vn0 = vvec[n0],           vn1 = vvec[n1];
    float cp0 = COVER ? covp[n0] : 0.f, cp1 = COVER ? covp[n1] : 0.f;
#pragma unroll
    for (int r = 0; r < 8; ++r) {
      float f00 = c00[r] + dp0, f10 = c10[r] + dp0;
      float f01 = c01[r] + dp1, f11 = c11[r] + dp1;
      if (COVER) {
        f00 += cvt[r] * cp0; f10 += cvt[8 + r] * cp0;
        f01 += cvt[r] * cp1; f11 += cvt[8 + r] * cp1;
      }
      acc[r]     += tanhf(f00) * vn0 + tanhf(f01) * vn1;
      acc[8 + r] += tanhf(f10) * vn0 + tanhf(f11) * vn1;
    }
  }
  // single reduction at the end: 16 LDS atomics per lane total
#pragma unroll
  for (int r = 0; r < 8; ++r) {
    int m = r + (half << 3);
    atomicAdd(&sscore[m], acc[r]);
    atomicAdd(&sscore[16 + m], acc[8 + r]);
  }
  __syncthreads();
  if (tid < 32 && (l0 + tid) < Ldim) scores[b * Ldim + l0 + tid] = sscore[tid];
}

// ===== kernel 5/6: masked softmax + renorm + coverage + context =====
__global__ void softmax_ctx_kernel(const float* __restrict__ scores,
                                   const float* __restrict__ mask,
                                   const float* __restrict__ mem, int Ldim,
                                   const float* __restrict__ cov_in,
                                   float* __restrict__ attn_out,
                                   float* __restrict__ cov_out,
                                   float* __restrict__ ctx_out) {
  int b = blockIdx.x, tid = threadIdx.x;
  __shared__ float sc[Ln];
  __shared__ float red[256];
  for (int l = tid; l < Ldim; l += 256) sc[l] = scores[b * Ldim + l];
  __syncthreads();
  float mx = -1e30f;
  for (int l = tid; l < Ldim; l += 256) mx = fmaxf(mx, sc[l]);
  red[tid] = mx; __syncthreads();
  for (int s = 128; s > 0; s >>= 1) { if (tid < s) red[tid] = fmaxf(red[tid], red[tid + s]); __syncthreads(); }
  mx = red[0]; __syncthreads();
  float sum = 0.f;
  for (int l = tid; l < Ldim; l += 256) { float e = __expf(sc[l] - mx); sc[l] = e; sum += e; }
  red[tid] = sum; __syncthreads();
  for (int s = 128; s > 0; s >>= 1) { if (tid < s) red[tid] += red[tid + s]; __syncthreads(); }
  float Z = red[0]; __syncthreads();
  float s2 = 0.f;
  for (int l = tid; l < Ldim; l += 256) { float a = (sc[l] / Z) * mask[b * Ldim + l]; sc[l] = a; s2 += a; }
  red[tid] = s2; __syncthreads();
  for (int s = 128; s > 0; s >>= 1) { if (tid < s) red[tid] += red[tid + s]; __syncthreads(); }
  float denom = red[0] + 1e-10f; __syncthreads();
  for (int l = tid; l < Ldim; l += 256) {
    float a = sc[l] / denom; sc[l] = a;
    if (attn_out) attn_out[b * Ldim + l] = a;
    if (cov_out)  cov_out[b * Ldim + l] = cov_in[b * Ldim + l] + a;
  }
  __syncthreads();
  for (int m = tid; m < Mn; m += 256) {
    float acc = 0.f;
    for (int l = 0; l < Ldim; ++l) acc += sc[l] * mem[((size_t)b * Ldim + l) * Mn + m];
    ctx_out[b * Mn + m] = acc;
  }
}

// ================= kernel 7: vd1 + p_gen =================
__global__ void vd1_pgen_kernel(const float* __restrict__ ctxW, const float* __restrict__ ctxS,
                                const float* __restrict__ hlast, const float* __restrict__ yemb,
                                const float* __restrict__ vd1_w, const float* __restrict__ vd1_b,
                                const float* __restrict__ pgen_w, const float* __restrict__ pgen_b,
                                float* __restrict__ hid, float* __restrict__ pgen_out) {
  int b = blockIdx.x, tid = threadIdx.x;
  __shared__ float xin[EMBn + Hn + 2 * Mn];
  __shared__ float red[256];
  for (int i = tid; i < Hn; i += 256) {
    xin[i]          = ctxW[b * Mn + i];
    xin[Mn + i]     = ctxS[b * Mn + i];
    xin[2 * Mn + i] = hlast[b * Hn + i];
  }
  if (tid < EMBn) xin[2 * Mn + Hn + tid] = yemb[b * EMBn + tid];
  __syncthreads();
  for (int j = tid; j < Hn; j += 256) {
    float acc = vd1_b[j];
    const float* w = &vd1_w[(size_t)j * (Hn + 2 * Mn)];
    for (int k = 0; k < Hn + 2 * Mn; ++k) acc += xin[k] * w[k];
    hid[b * Hn + j] = acc;
  }
  float p = 0.f;
  for (int k = tid; k < EMBn + Hn + 2 * Mn; k += 256) p += xin[k] * pgen_w[k];
  red[tid] = p; __syncthreads();
  for (int s = 128; s > 0; s >>= 1) { if (tid < s) red[tid] += red[tid + s]; __syncthreads(); }
  if (tid == 0) pgen_out[b] = 1.f / (1.f + __expf(-(red[0] + pgen_b[0])));
}

// ====== kernel 8: vocab logits GEMM (bf16 WMMA, packed weights) ======
// 32 batch rows per block; each wave: 2 N-tiles x 2 M-tiles = 4 WMMA chains.
__global__ void vocab_gemm_wmma(const float* __restrict__ hid,
                                const unsigned int* __restrict__ Wp,  // packed [256][V]
                                const float* __restrict__ bias,
                                float* __restrict__ logits) {
  int b0 = blockIdx.y << 5;                  // 4 blocks of 32 batch rows
  int tid = threadIdx.x, wave = tid >> 5, lane = tid & 31;
  __shared__ unsigned int slab[32][257];
  for (int i = tid; i < 32 * 256; i += 128) {
    int r = i >> 8, p = i & 255;
    const float* src = &hid[(size_t)(b0 + r) * Hn + 2 * p];
    slab[r][p] = pack2bf(src[0], src[1]);
  }
  __syncthreads();
  int nt0 = blockIdx.x * 4 + wave;           // 0..1563
  int nt1 = nt0 + 1564;                      // tail clamped to duplicate (store-guarded)
  bool nt1ok = (nt1 < NTv);
  int n0 = nt0 << 4;
  int n1 = (nt1ok ? nt1 : nt0) << 4;
  fvec8 c00 = FZ8, c10 = FZ8, c01 = FZ8, c11 = FZ8;
#pragma unroll 2
  for (int kc = 0; kc < 16; ++kc) {
    bfvec16 bf0 = load_b_packed(Wp, n0, kc * 32, Vn);
    bfvec16 bf1 = load_b_packed(Wp, n1, kc * 32, Vn);
    bfvec16 a0  = load_a_packed(slab, 0,  kc * 32);
    bfvec16 a1  = load_a_packed(slab, 16, kc * 32);
    c00 = WMMA_BF16(a0, bf0, c00);
    c10 = WMMA_BF16(a1, bf0, c10);
    c01 = WMMA_BF16(a0, bf1, c01);
    c11 = WMMA_BF16(a1, bf1, c11);
  }
  int cn0 = n0 + (lane & 15), cn1 = n1 + (lane & 15);
  float bv0 = bias[cn0], bv1 = bias[cn1];
  int half = lane >> 4;
#pragma unroll
  for (int r = 0; r < 8; ++r) {
    int m = r + (half << 3);
    logits[(size_t)(b0 + m) * Vn + cn0]      = c00[r] + bv0;
    logits[(size_t)(b0 + 16 + m) * Vn + cn0] = c10[r] + bv0;
    if (nt1ok) {
      logits[(size_t)(b0 + m) * Vn + cn1]      = c01[r] + bv1;
      logits[(size_t)(b0 + 16 + m) * Vn + cn1] = c11[r] + bv1;
    }
  }
}

// ===== kernel 9: vocab softmax * p_gen -> final_dist[:V]; zero OOV slots =====
__global__ void vocab_softmax_kernel(const float* __restrict__ logits,
                                     const float* __restrict__ pgen,
                                     float* __restrict__ final_dist) {
  int b = blockIdx.x, tid = threadIdx.x;
  __shared__ float red[256];
  const float* lrow = &logits[(size_t)b * Vn];
  float mx = -1e30f;
  for (int v = tid; v < Vn; v += 256) mx = fmaxf(mx, lrow[v]);
  red[tid] = mx; __syncthreads();
  for (int s = 128; s > 0; s >>= 1) { if (tid < s) red[tid] = fmaxf(red[tid], red[tid + s]); __syncthreads(); }
  mx = red[0]; __syncthreads();
  float sum = 0.f;
  for (int v = tid; v < Vn; v += 256) sum += __expf(lrow[v] - mx);
  red[tid] = sum; __syncthreads();
  for (int s = 128; s > 0; s >>= 1) { if (tid < s) red[tid] += red[tid + s]; __syncthreads(); }
  float Z = red[0];
  float pg = pgen[b];
  float* frow = &final_dist[(size_t)b * VPn];
  for (int v = tid; v < Vn; v += 256) frow[v] = pg * __expf(lrow[v] - mx) / Z;
  for (int v = Vn + tid; v < VPn; v += 256) frow[v] = 0.f;
}

// ===== kernel 10: scatter-add of copy distribution =====
__global__ void scatter_kernel(const int* __restrict__ src_oov,
                               const float* __restrict__ attn,
                               const float* __restrict__ pgen,
                               float* __restrict__ final_dist) {
  int b = blockIdx.x, tid = threadIdx.x;
  float w = 1.f - pgen[b];
  for (int l = tid; l < Ln; l += 256) {
    int idx = src_oov[b * Ln + l];
    atomicAdd(&final_dist[(size_t)b * VPn + idx], w * attn[b * Ln + l]);
  }
}

// ================= launcher =================
extern "C" void kernel_launch(void* const* d_in, const int* in_sizes, int n_in,
                              void* d_out, int out_size, void* d_ws, size_t ws_size,
                              hipStream_t stream) {
  const float* h          = (const float*)d_in[0];
  const float* wmem       = (const float*)d_in[1];
  const float* smem       = (const float*)d_in[2];
  const float* wmask      = (const float*)d_in[3];
  const float* smask      = (const float*)d_in[4];
  const float* coverage   = (const float*)d_in[5];
  const float* emb_table  = (const float*)d_in[6];
  const float* gw_ih      = (const float*)d_in[7];
  const float* gw_hh      = (const float*)d_in[8];
  const float* gb_ih      = (const float*)d_in[9];
  const float* gb_hh      = (const float*)d_in[10];
  const float* w_mem_proj = (const float*)d_in[11];
  const float* w_dec_w    = (const float*)d_in[12];
  const float* w_dec_b    = (const float*)d_in[13];
  const float* w_v        = (const float*)d_in[14];
  const float* w_covp     = (const float*)d_in[15];
  const float* s_mem_proj = (const float*)d_in[16];
  const float* s_dec_w    = (const float*)d_in[17];
  const float* s_dec_b    = (const float*)d_in[18];
  const float* s_v        = (const float*)d_in[19];
  const float* pgen_w     = (const float*)d_in[20];
  const float* pgen_b     = (const float*)d_in[21];
  const float* vd1_w      = (const float*)d_in[22];
  const float* vd1_b      = (const float*)d_in[23];
  const float* vd2_w      = (const float*)d_in[24];
  const float* vd2_b      = (const float*)d_in[25];
  const int*   y          = (const int*)d_in[26];
  const int*   src_oov    = (const int*)d_in[27];

  float* out = (float*)d_out;
  float* ws  = (float*)d_ws;
  unsigned int* wmpk  = (unsigned int*)(ws + WS_WMPK);
  unsigned int* smpk  = (unsigned int*)(ws + WS_SMPK);
  unsigned int* vd2pk = (unsigned int*)(ws + WS_VD2PK);

  float* final_dist = out + OUT_FINAL;
  float* h_next     = out + OUT_HNEXT;   // doubles as h_last
  float* ctxW       = out + OUT_CTXW;
  float* attnW      = out + OUT_ATTNW;
  float* pgen       = out + OUT_PGEN;
  float* covW       = out + OUT_COVW;

  // 0) pack weights to bf16-pair layout (independent of everything else)
  pack512_kernel<<<64,   256, 0, stream>>>(w_mem_proj, wmpk, Hn);
  pack512_kernel<<<64,   256, 0, stream>>>(s_mem_proj, smpk, Hn);
  pack512_kernel<<<4096, 256, 0, stream>>>(vd2_w, vd2pk, Vn);
  // 1) embedding + GRU step -> h_last
  gru_kernel<<<Bn, 256, 0, stream>>>(emb_table, y, h, gw_ih, gw_hh, gb_ih, gb_hh,
                                     h_next, ws + WS_YEMB);
  // 2) decoder projections
  decproj_kernel<<<dim3(Bn, 2), 256, 0, stream>>>(h_next, w_dec_w, w_dec_b,
                                                  s_dec_w, s_dec_b,
                                                  ws + WS_DPROJW, ws + WS_DPROJS);
  // 3) word attention scores (dominant GEMM)
  attn_score_wmma<true><<<dim3((Ln + 31) / 32, Bn), 128, 0, stream>>>(
      wmem, Ln, wmpk, ws + WS_DPROJW, w_v, w_covp, coverage, ws + WS_SCORESW);
  // 4) sentence attention scores
  attn_score_wmma<false><<<dim3((Sn + 31) / 32, Bn), 128, 0, stream>>>(
      smem, Sn, smpk, ws + WS_DPROJS, s_v, nullptr, nullptr, ws + WS_SCORESS);
  // 5) word softmax + coverage + context
  softmax_ctx_kernel<<<Bn, 256, 0, stream>>>(ws + WS_SCORESW, wmask, wmem, Ln,
                                             coverage, attnW, covW, ctxW);
  // 6) sentence softmax + context
  softmax_ctx_kernel<<<Bn, 256, 0, stream>>>(ws + WS_SCORESS, smask, smem, Sn,
                                             nullptr, nullptr, nullptr, ws + WS_CTXS);
  // 7) vd1 + p_gen
  vd1_pgen_kernel<<<Bn, 256, 0, stream>>>(ctxW, ws + WS_CTXS, h_next, ws + WS_YEMB,
                                          vd1_w, vd1_b, pgen_w, pgen_b,
                                          ws + WS_HID, pgen);
  // 8) vocab logits GEMM: 32-row blocks, 2 N-tiles per wave
  vocab_gemm_wmma<<<dim3(391, Bn / 32), 128, 0, stream>>>(
      ws + WS_HID, vd2pk, vd2_b, ws + WS_LOGITS);
  // 9) vocab softmax * p_gen + zero OOV tail
  vocab_softmax_kernel<<<Bn, 256, 0, stream>>>(ws + WS_LOGITS, pgen, final_dist);
  // 10) scatter-add copy distribution
  scatter_kernel<<<Bn, 256, 0, stream>>>(src_oov, attnW, pgen, final_dist);
}
